// MultiHeadSelfAttention_6743098655379
// MI455X (gfx1250) — matrix operand order repaired and checked by reading, not compiled, not run on hardware
//
#include <hip/hip_runtime.h>

// ---------------------------------------------------------------------------
// MHA for MI455X (gfx1250, wave32, WMMA 16x16x32 f16 -> f32 accum)
//   x:[4,2048,1024] f32, W_qkv:[1024,3072], b_qkv:[3072], W_out:[1024,1024],
//   b_out:[1024]  ->  out:[4,2048,1024] f32
// ---------------------------------------------------------------------------

typedef _Float16 h16v __attribute__((ext_vector_type(16)));
typedef _Float16 h8v  __attribute__((ext_vector_type(8)));
typedef float    f8v  __attribute__((ext_vector_type(8)));

static __device__ __forceinline__ f8v wmma32(h16v a, h16v b, f8v c) {
  // (neg_a, A, neg_b, B, c_mod, C, reuse_a, reuse_b)
  return __builtin_amdgcn_wmma_f32_16x16x32_f16(false, a, false, b, (short)0, c,
                                                false, false);
}

static __device__ __forceinline__ f8v zero8() {
  f8v z;
#pragma unroll
  for (int i = 0; i < 8; ++i) z[i] = 0.0f;
  return z;
}

// A fragment (16x32, f16) from a row-major matrix (ld halves per row).
// Lane L -> row L%16; with g=L/16, elements 0..15 hold K=16g..16g+15
// => one contiguous 16-half run per lane (matches B fragments).
static __device__ __forceinline__ h16v load_a_frag(const _Float16* base, int ld,
                                                   int lane) {
  const int r = lane & 15, g = lane >> 4;
  const _Float16* p = base + (size_t)r * ld + 16 * g;
  return *(const h16v*)p;
}

// B fragment (32x16, f16). Source is "N-major": row n holds the 32 K-values
// contiguously. Lane L -> column n = L%16, K-run starts at 16*(L/16):
// one contiguous 16-half (32B) load per lane.
static __device__ __forceinline__ h16v load_b_frag(const _Float16* base, int ld,
                                                   int lane) {
  const _Float16* p = base + (size_t)(lane & 15) * ld + 16 * (lane >> 4);
  return *(const h16v*)p;
}

// ---------------------------------------------------------------------------
__global__ __launch_bounds__(256) void mha_cvt_f32_f16(const float* __restrict__ in,
                                                       _Float16* __restrict__ out,
                                                       int n) {
  int i = blockIdx.x * blockDim.x + threadIdx.x;
  const int stride = gridDim.x * blockDim.x;
  for (; i < n; i += stride) out[i] = (_Float16)in[i];
}

// ---------------------------------------------------------------------------
// GEMM: Out[M,N] = A[M,K] (f16, row-major) * B[K,N] (f16, row-major) + bias.
// Block: 256 thr = 8 waves; block tile 256(M) x 64(N); wave tile 32 x 64
// (2 A frags x 4 B frags = 8 wmma per k-step, B frags reused across A frags).
// B k-slab staged transposed into LDS (ldsBT[n][k]) for contiguous B frags.
template <bool F16OUT, bool QSCALE>
__global__ __launch_bounds__(256) void mha_gemm_wmma(
    const _Float16* __restrict__ A, int lda,
    const _Float16* __restrict__ B, int ldb,
    const float* __restrict__ bias,
    void* __restrict__ Out, int ldc, int K) {
  __shared__ _Float16 ldsBT[64][32];
  const int tid = threadIdx.x;
  const int lane = tid & 31;
  const int wave = tid >> 5;
  const int rowBase = blockIdx.x * 256 + wave * 32;
  const int colBase = blockIdx.y * 64;

  f8v acc[2][4];
#pragma unroll
  for (int m = 0; m < 2; ++m)
#pragma unroll
    for (int j = 0; j < 4; ++j) acc[m][j] = zero8();

  const int kThread = tid >> 3;       // 0..31
  const int nThread = (tid & 7) * 8;  // 0..56

  for (int k0 = 0; k0 < K; k0 += 32) {
    __syncthreads();
    {  // stage 32(K) x 64(N) slab of B transposed into LDS
      const _Float16* bp = B + (size_t)(k0 + kThread) * ldb + colBase + nThread;
      h8v bv = *(const h8v*)bp;
#pragma unroll
      for (int i = 0; i < 8; ++i) ldsBT[nThread + i][kThread] = bv[i];
    }
    __syncthreads();
    h16v a[2];
#pragma unroll
    for (int m = 0; m < 2; ++m) {
      const _Float16* arow = A + (size_t)(rowBase + 16 * m) * lda + k0;
      a[m] = load_a_frag(arow, lda, lane);
      if (k0 + 32 < K) __builtin_prefetch(arow + 32, 0, 1);
    }
#pragma unroll
    for (int j = 0; j < 4; ++j) {
      h16v b = load_b_frag(&ldsBT[j * 16][0], 32, lane);
#pragma unroll
      for (int m = 0; m < 2; ++m) acc[m][j] = wmma32(a[m], b, acc[m][j]);
    }
  }

  const int g = lane >> 4;
#pragma unroll
  for (int m = 0; m < 2; ++m) {
#pragma unroll
    for (int j = 0; j < 4; ++j) {
      const int col = colBase + j * 16 + (lane & 15);
      const float bval = bias[col];
#pragma unroll
      for (int v = 0; v < 8; ++v) {
        const int row = rowBase + 16 * m + v + 8 * g;  // C: M = v + 8*(L/16)
        float val = acc[m][j][v] + bval;
        if (QSCALE) {           // fold 1/sqrt(Hd)=0.125 into stored Q columns
          if (col < 1024) val *= 0.125f;
        }
        if (F16OUT)
          ((_Float16*)Out)[(size_t)row * ldc + col] = (_Float16)val;
        else
          ((float*)Out)[(size_t)row * ldc + col] = val;
      }
    }
  }
}

// ---------------------------------------------------------------------------
// Flash attention. qkv:[B*2048, 3072] f16 (Q|K|V each 1024 cols, head h at
// h*64). One block per (b, h, 128 q-rows); wave owns 16 q-rows. Keys in
// chunks of 64: S(16x64) = Q(16x64) Kᵀ (8 wmma), online softmax (reductions
// amortized over 64 keys), O += P V (8 wmma). Scale pre-folded into Q.
__global__ __launch_bounds__(256) void mha_flash_attn(
    const _Float16* __restrict__ qkv, _Float16* __restrict__ attnOut) {
  __shared__ _Float16 ldsVT[64][64];    // Vᵀ[hd][key] of current chunk (8 KB)
  __shared__ _Float16 ldsP[8][16][64];  // per-wave P tile, C->A relayout (16KB)
  const int tid = threadIdx.x;
  const int lane = tid & 31;
  const int wave = tid >> 5;
  const int h = blockIdx.y;
  const int b = blockIdx.z;
  const int qBase = blockIdx.x * 128 + wave * 16;
  const int ld = 3072;

  const _Float16* Qb = qkv + (size_t)(b * 2048 + qBase) * ld + h * 64;
  const _Float16* Kb = qkv + (size_t)(b * 2048) * ld + 1024 + h * 64;
  const _Float16* Vb = qkv + (size_t)(b * 2048) * ld + 2048 + h * 64;

  const h16v qa0 = load_a_frag(Qb, ld, lane);        // hd 0..31
  const h16v qa1 = load_a_frag(Qb + 32, ld, lane);   // hd 32..63

  f8v oacc[4];
#pragma unroll
  for (int t = 0; t < 4; ++t) oacc[t] = zero8();
  float mi[8], li[8];
#pragma unroll
  for (int v = 0; v < 8; ++v) { mi[v] = -1e30f; li[v] = 0.0f; }

  const int keyT = tid >> 2;        // 0..63
  const int hdT = (tid & 3) * 16;   // 0,16,32,48

  for (int kc = 0; kc < 2048; kc += 64) {
    __syncthreads();  // previous chunk's ldsVT reads are done
    {                 // cooperatively stage V chunk transposed: ldsVT[hd][key]
      const _Float16* vp = Vb + (size_t)(kc + keyT) * ld + hdT;
      h16v vv = *(const h16v*)vp;
#pragma unroll
      for (int i = 0; i < 16; ++i) ldsVT[hdT + i][keyT] = vv[i];
    }

    // S = Q Kᵀ for four 16-key sub-tiles; K rows are per-lane contiguous runs
    f8v s[4];
#pragma unroll
    for (int st = 0; st < 4; ++st) {
      const _Float16* kp = Kb + (size_t)(kc + st * 16) * ld;
      h16v kb0 = load_b_frag(kp, ld, lane);       // hd 0..31
      h16v kb1 = load_b_frag(kp + 32, ld, lane);  // hd 32..63
      f8v z = zero8();
      z = wmma32(qa0, kb0, z);
      z = wmma32(qa1, kb1, z);
      s[st] = z;
    }

    // Online softmax; a C-tile row lives in a 16-lane half-group at elem v.
#pragma unroll
    for (int v = 0; v < 8; ++v) {
      float smax = fmaxf(fmaxf(s[0][v], s[1][v]), fmaxf(s[2][v], s[3][v]));
#pragma unroll
      for (int d = 1; d < 16; d <<= 1)
        smax = fmaxf(smax, __shfl_xor(smax, d, 32));
      const float mnew = fmaxf(mi[v], smax);
      const float alpha = __expf(mi[v] - mnew);
      float p[4];
      float rs = 0.0f;
#pragma unroll
      for (int st = 0; st < 4; ++st) {
        p[st] = __expf(s[st][v] - mnew);
        rs += p[st];
      }
#pragma unroll
      for (int d = 1; d < 16; d <<= 1) rs += __shfl_xor(rs, d, 32);
      li[v] = li[v] * alpha + rs;
      mi[v] = mnew;
#pragma unroll
      for (int t = 0; t < 4; ++t) oacc[t][v] *= alpha;
      const int prow = v + 8 * (lane >> 4);
#pragma unroll
      for (int st = 0; st < 4; ++st)
        ldsP[wave][prow][st * 16 + (lane & 15)] = (_Float16)p[st];
    }
    __syncthreads();  // ldsVT ready for everyone

    // O += P(16x64) @ V(64x64): 2 P A-frags (keys 0..31 / 32..63) x 4 hd tiles
    const h16v pa0 = load_a_frag(&ldsP[wave][0][0], 64, lane);
    const h16v pa1 = load_a_frag(&ldsP[wave][0][0] + 32, 64, lane);
#pragma unroll
    for (int t = 0; t < 4; ++t) {
      const _Float16* vrow = &ldsVT[t * 16 + (lane & 15)][0];
      const h16v vb0 = *(const h16v*)(vrow + 16 * (lane >> 4));
      const h16v vb1 = *(const h16v*)(vrow + 32 + 16 * (lane >> 4));
      oacc[t] = wmma32(pa0, vb0, oacc[t]);
      oacc[t] = wmma32(pa1, vb1, oacc[t]);
    }
  }

  // Normalize and emit f16 [B*N, 1024] for the output projection
#pragma unroll
  for (int v = 0; v < 8; ++v) {
    const float inv = 1.0f / li[v];
    const int q = qBase + v + 8 * (lane >> 4);
#pragma unroll
    for (int t = 0; t < 4; ++t) {
      attnOut[(size_t)(b * 2048 + q) * 1024 + h * 64 + t * 16 + (lane & 15)] =
          (_Float16)(oacc[t][v] * inv);
    }
  }
}

// ---------------------------------------------------------------------------
extern "C" void kernel_launch(void* const* d_in, const int* in_sizes, int n_in,
                              void* d_out, int out_size, void* d_ws,
                              size_t ws_size, hipStream_t stream) {
  (void)in_sizes; (void)n_in; (void)out_size; (void)ws_size;
  const float* x    = (const float*)d_in[0];  // [4,2048,1024]
  const float* Wqkv = (const float*)d_in[1];  // [1024,3072]
  const float* bqkv = (const float*)d_in[2];  // [3072]
  const float* Wout = (const float*)d_in[3];  // [1024,1024]
  const float* bout = (const float*)d_in[4];  // [1024]

  char* ws = (char*)d_ws;
  const size_t MB = 1024 * 1024;
  _Float16* xh    = (_Float16*)(ws + 0);        // 16 MB
  _Float16* wqkvh = (_Float16*)(ws + 16 * MB);  //  6 MB
  _Float16* wouth = (_Float16*)(ws + 22 * MB);  //  2 MB
  _Float16* qkvh  = (_Float16*)(ws + 24 * MB);  // 48 MB  [8192,3072] f16
  _Float16* attnh = (_Float16*)(ws + 72 * MB);  // 16 MB  [8192,1024] f16

  mha_cvt_f32_f16<<<1024, 256, 0, stream>>>(x, xh, 4 * 2048 * 1024);
  mha_cvt_f32_f16<<<512, 256, 0, stream>>>(Wqkv, wqkvh, 1024 * 3072);
  mha_cvt_f32_f16<<<256, 256, 0, stream>>>(Wout, wouth, 1024 * 1024);

  // QKV = x @ W_qkv + b_qkv, Q scaled by 0.125, stored f16
  mha_gemm_wmma<true, true><<<dim3(32, 48), 256, 0, stream>>>(
      xh, 1024, wqkvh, 3072, bqkv, (void*)qkvh, 3072, 1024);

  // Flash attention over all (b, h)
  mha_flash_attn<<<dim3(16, 16, 4), 256, 0, stream>>>(qkvh, attnh);

  // out = attn @ W_out + b_out, f32 to d_out
  mha_gemm_wmma<false, false><<<dim3(32, 16), 256, 0, stream>>>(
      attnh, 1024, wouth, 1024, bout, d_out, 1024, 1024);
}
